// GCDN_2757369004725
// MI455X (gfx1250) — matrix operand order repaired and checked by reading, not compiled, not run on hardware
//
#include <hip/hip_runtime.h>
#include <hip/hip_bf16.h>
#include <math.h>

// ---------------------------------------------------------------------------
// GCDN edge-conditioned graph conv for MI455X (gfx1250, wave32, WMMA).
//
//   1) prep_h:  h[4096x44] f32 -> h16[4096x64] f16 (zero pad), sq norms.
//   2) pack_w:  weights (+ bias folded into K=44 homogeneous row) pre-swizzled
//               into B-fragment order, contiguous k = 16*hi + s (lane-contiguous 32B).
//   3) knn:     fused Gram tile (v_wmma_f32_16x16x32_f16) + distance +
//               masked top-25 selection. Never materializes the 64MB D.
//   4) edge:    8 waves/block, one node (16 edges) per wave. theta1/theta2
//               fragment buffers (135KB) staged into LDS once per block via
//               GLOBAL_LOAD_ASYNC_TO_LDS_B128 (ASYNCcnt), shared by all 8
//               waves -> 8x less L2 weight traffic. All GEMMs via WMMA with
//               fused rank contractions in fragment epilogues.
// ---------------------------------------------------------------------------

#ifndef USE_ASYNC_LDS
#define USE_ASYNC_LDS 1
#endif

typedef __attribute__((ext_vector_type(16))) _Float16 v16h;
typedef __attribute__((ext_vector_type(8)))  float    v8f;

#define N_NODES 4096
#define F_INN   44
#define F_OUTN  44
#define KNN     16
#define RANKN   11
#define KPAD    64      // K dim padded (44 data + 1 homogeneous + zeros)
#define FPAD    48      // row stride for 16xF LDS tiles

// ---- ws layout (bytes; total ~962KB, assumes ws_size >= 1MB) ----
#define WS_H16    0          // 4096*64*2   = 524288
#define WS_SQ     524288     // 4096*4      = 16384
#define WS_NBR    540672     // 65536*4     = 262144
#define WS_CTR    802816     // 4096*4      = 16384
#define WS_WLABF  819200     // 1*3*2*512*2 = 6144
#define WS_WTHLF  825344     // 1*1*2*512*2 = 2048
#define WS_WTH1F  827392     // 11*3*2*512*2= 67584   (th1 then th2 contiguous)
#define WS_WTH2F  894976     // 11*3*2*512*2= 67584

#define WTHF_HALVES   33792            // halves per theta fragment buffer
#define WTHF_BYTES    67584
#define WSTAGE_BYTES  (2 * WTHF_BYTES) // 135168, staged to LDS per block

// edge kernel LDS: staged weights + 8 per-wave scratch slabs
#define PW_X1    0
#define PW_LBL   3072
#define PW_LAB   6144
#define PW_THL   9216
#define PW_T     10240
#define PW_DL    11008
#define PW_NBR   11072
#define PW_CTR   11136
#define PER_WAVE 11264
#define SMEM_EDGE (WSTAGE_BYTES + 8 * PER_WAVE)   // 225280 <= 320KB/WGP

__device__ __forceinline__ v8f wmma_f16(v16h a, v16h b, v8f c) {
  return __builtin_amdgcn_wmma_f32_16x16x32_f16(
      /*neg_a=*/false, a, /*neg_b=*/false, b,
      /*c_mod=*/(short)0, c, /*reuse_a=*/false, /*reuse_b=*/false);
}

// A-fragment loader for a 64-wide f16 row (contiguous k order, same as B):
//   lane: M = lane&15 ; per VGPR v: k = kc*32 + hi*16 + 2*v
__device__ __forceinline__ v16h load_a_row64(const _Float16* row, int hi, int kc) {
  v16h a;
  const _Float16* p0 = row + kc * 32 + hi * 16;
  const _Float16* p1 = p0 + 8;
#pragma unroll
  for (int i = 0; i < 8; ++i) { a[i] = p0[i]; a[8 + i] = p1[i]; }
  return a;
}

// Build an A fragment from an f32 LDS tile [16][FPAD], with value `homv`
// injected at k==44 (homogeneous coordinate -> folds the bias row in).
__device__ __forceinline__ v16h make_a_from_lds(const float* mat, int M, int hi,
                                                int kc, float homv) {
  v16h a;
#pragma unroll
  for (int v = 0; v < 8; ++v) {
    int k0 = kc * 32 + hi * 16 + 2 * v;
    int k1 = k0 + 1;
    float f0 = (k0 == 44) ? homv : (k0 < 44 ? mat[M * FPAD + k0] : 0.0f);
    float f1 = (k1 == 44) ? homv : (k1 < 44 ? mat[M * FPAD + k1] : 0.0f);
    a[2 * v]     = (_Float16)f0;
    a[2 * v + 1] = (_Float16)f1;
  }
  return a;
}

// Load a pre-swizzled B fragment (slot = 512 halves, lane-contiguous 32B).
// Two 16B pieces -> works for 16B-aligned LDS or global sources.
__device__ __forceinline__ v16h load_b_frag(const _Float16* base, int slot, int lane) {
  const float4* p = (const float4*)(base + (size_t)slot * 512 + (size_t)lane * 16);
  union { float4 f[2]; v16h h; } u;
  u.f[0] = p[0];
  u.f[1] = p[1];
  return u.h;
}

// ---------------------------------------------------------------------------
// Kernel 1: pad/convert h to f16, compute squared norms.
// ---------------------------------------------------------------------------
__global__ void prep_h_kernel(const float* __restrict__ h,
                              _Float16* __restrict__ h16,
                              float* __restrict__ sq) {
  int i = blockIdx.x * blockDim.x + threadIdx.x;
  if (i >= N_NODES) return;
  float s = 0.0f;
#pragma unroll
  for (int f = 0; f < KPAD; ++f) {
    float v = (f < F_INN) ? h[i * F_INN + f] : 0.0f;
    h16[i * KPAD + f] = (_Float16)v;
    s += v * v;
  }
  sq[i] = s;
}

// ---------------------------------------------------------------------------
// Kernel 2: pack a weight matrix (+ bias at k==44) into B-fragment order.
// ---------------------------------------------------------------------------
__global__ void pack_w_kernel(const float* __restrict__ W,
                              const float* __restrict__ B,
                              int ldW, int rcount, int ntc, int colsPerR,
                              _Float16* __restrict__ out) {
  int tid = blockIdx.x * blockDim.x + threadIdx.x;
  int total = rcount * ntc * 2 * 256;
  if (tid >= total) return;
  int dw   = tid & 255;
  int slot = tid >> 8;
  int kc   = slot & 1;
  int ntr  = slot >> 1;
  int nt   = ntr % ntc;
  int r    = ntr / ntc;
  int lane = dw >> 3;
  int v    = dw & 7;
  int hi   = lane >> 4;
  int n    = lane & 15;
  int f    = nt * 16 + n;
  int k0   = kc * 32 + hi * 16 + 2 * v;

  float v0 = 0.0f, v1 = 0.0f;
  if (f < colsPerR) {
    int col = r * colsPerR + f;
    if (k0 < 44)        v0 = W[k0 * ldW + col];
    else if (k0 == 44)  v0 = B[col];
    int k1 = k0 + 1;
    if (k1 < 44)        v1 = W[k1 * ldW + col];
    else if (k1 == 44)  v1 = B[col];
  }
  out[(size_t)tid * 2 + 0] = (_Float16)v0;
  out[(size_t)tid * 2 + 1] = (_Float16)v1;
}

// ---------------------------------------------------------------------------
// Kernel 3: fused Gram + distance + top-25 kNN. One wave per 16 rows.
// ---------------------------------------------------------------------------
__global__ void __launch_bounds__(32)
knn_kernel(const _Float16* __restrict__ h16,
           const float* __restrict__ sq,
           const float* __restrict__ mask,
           int* __restrict__ nbr, int* __restrict__ ctr) {
  __shared__ float s_d[16][25];
  __shared__ int   s_i[16][25];
  __shared__ float s_tile[16][17];

  const int lane = threadIdx.x;
  const int rblk = blockIdx.x;
  const int hi   = lane >> 4;
  const int M    = lane & 15;

  for (int idx = lane; idx < 16 * 25; idx += 32) {
    s_d[idx / 25][idx % 25] = 1.0e30f;
    s_i[idx / 25][idx % 25] = 0;
  }
  __syncthreads();

  const _Float16* arow = h16 + (size_t)(rblk * 16 + M) * KPAD;
  v16h a0 = load_a_row64(arow, hi, 0);
  v16h a1 = load_a_row64(arow, hi, 1);

  float sqi[8];
#pragma unroll
  for (int v = 0; v < 8; ++v) sqi[v] = sq[rblk * 16 + v + hi * 8];

  for (int jt = 0; jt < N_NODES / 16; ++jt) {
    int j = jt * 16 + (lane & 15);
    // B fragment: 32x16 f16, lane n = lane&15, k = kc*32 + hi*16 + 2v{,+1}
    v16h b0 = *(const v16h*)(h16 + (size_t)j * KPAD + hi * 16);
    v16h b1 = *(const v16h*)(h16 + (size_t)j * KPAD + 32 + hi * 16);
    v8f c = {};
    c = wmma_f16(a0, b0, c);
    c = wmma_f16(a1, b1, c);
    float sqj = sq[j];
    if (jt + 1 < N_NODES / 16) {   // prefetch next mask tile rows
      __builtin_prefetch(&mask[(size_t)(rblk * 16 + M) * N_NODES + (jt + 1) * 16], 0, 0);
    }
#pragma unroll
    for (int v = 0; v < 8; ++v) {
      int i = rblk * 16 + v + hi * 8;
      float d = fabsf(sqi[v] + sqj - 2.0f * c[v]);
      d *= mask[(size_t)i * N_NODES + j];
      if (i == j) d -= 2.0f;
      s_tile[v + hi * 8][lane & 15] = d;
    }
    __syncthreads();
    if (lane < 16) {
      for (int cc = 0; cc < 16; ++cc) {
        float d = s_tile[lane][cc];
        if (d < s_d[lane][24]) {
          int pos = 24;
          while (pos > 0 && s_d[lane][pos - 1] > d) {
            s_d[lane][pos] = s_d[lane][pos - 1];
            s_i[lane][pos] = s_i[lane][pos - 1];
            --pos;
          }
          s_d[lane][pos] = d;
          s_i[lane][pos] = jt * 16 + cc;
        }
      }
    }
    __syncthreads();
  }

  if (lane < 16) {
    int row = rblk * 16 + lane;
    ctr[row] = s_i[lane][0];                  // rank 0 == self
#pragma unroll
    for (int k = 0; k < KNN; ++k) nbr[row * KNN + k] = s_i[lane][9 + k];
  }
}

// ---------------------------------------------------------------------------
// Kernel 4: edge MLPs + low-rank conv + weighted mean.
// 8 waves/block, one node per wave; theta weights staged in LDS per block.
// wthf = packed th1 fragments, immediately followed by th2 (contiguous in ws).
// ---------------------------------------------------------------------------
__global__ void __launch_bounds__(256)
edge_kernel(const float* __restrict__ h,
            const int* __restrict__ nbr, const int* __restrict__ ctr,
            const _Float16* __restrict__ wlabf,
            const _Float16* __restrict__ wthlf,
            const _Float16* __restrict__ wthf,
            float* __restrict__ out) {
  extern __shared__ char smem[];
  _Float16* w1 = (_Float16*)smem;                       // th1 frags (LDS)
  _Float16* w2 = (_Float16*)(smem + WTHF_BYTES);        // th2 frags (LDS)

  const int tid  = threadIdx.x;
  const int lane = tid & 31;
  const int wid  = tid >> 5;
  const int node = blockIdx.x * 8 + wid;
  const int hi   = lane >> 4;
  const int nlo  = lane & 15;

  // ---- stage theta fragment buffers into LDS (block-cooperative) ----
  {
    const int nchunk = WSTAGE_BYTES / 16;               // 8448 x B128
#if USE_ASYNC_LDS
    for (int c = tid; c < nchunk; c += 256) {
      unsigned long long ga = (unsigned long long)(uintptr_t)((const char*)wthf + (size_t)c * 16);
      unsigned int la = (unsigned int)(uintptr_t)(smem + (size_t)c * 16);  // low 32b = LDS addr
      asm volatile("global_load_async_to_lds_b128 %0, %1, off"
                   :: "v"(la), "v"(ga) : "memory");
    }
    asm volatile("s_wait_asynccnt 0" ::: "memory");
#else
    for (int c = tid; c < nchunk; c += 256) {
      *(float4*)(smem + (size_t)c * 16) =
          *(const float4*)((const char*)wthf + (size_t)c * 16);
    }
#endif
  }

  // ---- per-wave scratch slabs ----
  char* wm = smem + WSTAGE_BYTES + (size_t)wid * PER_WAVE;
  float* s_x1  = (float*)(wm + PW_X1);
  float* s_lbl = (float*)(wm + PW_LBL);
  float* s_lab = (float*)(wm + PW_LAB);
  float* s_thl = (float*)(wm + PW_THL);
  float* s_t   = (float*)(wm + PW_T);
  float* s_dl  = (float*)(wm + PW_DL);
  int*   s_nbr = (int*)  (wm + PW_NBR);
  int*   s_ctr = (int*)  (wm + PW_CTR);

  if (lane < 16) s_nbr[lane] = nbr[node * KNN + lane];
  if (lane == 0) s_ctr[0] = ctr[node];
  for (int idx = lane; idx < 16 * FPAD; idx += 32) {
    s_x1[idx] = 0.0f; s_lbl[idx] = 0.0f;
  }
  __syncthreads();   // also covers weight staging completion

  // ---- gather + labels ----
  const int cidx = s_ctr[0];
  for (int idx = lane; idx < 16 * F_INN; idx += 32) {
    int k = idx / F_INN, f = idx % F_INN;
    float xn = h[(size_t)s_nbr[k] * F_INN + f];
    float xc = h[(size_t)cidx * F_INN + f];
    s_x1[k * FPAD + f]  = xn;
    s_lbl[k * FPAD + f] = xn - xc;
  }
  __syncthreads();
  if (lane < 16) {
    float s = 0.0f;
#pragma unroll
    for (int f = 0; f < F_INN; ++f) { float d = s_lbl[lane * FPAD + f]; s += d * d; }
    s_dl[lane] = s;
  }
  __syncthreads();

  v16h a_lbl0 = make_a_from_lds(s_lbl, nlo, hi, 0, 1.0f);
  v16h a_lbl1 = make_a_from_lds(s_lbl, nlo, hi, 1, 1.0f);

  // GEMM 1: lab = leakyrelu(labels @ Wlab + b)
#pragma unroll
  for (int nt = 0; nt < 3; ++nt) {
    v16h b0 = load_b_frag(wlabf, nt * 2 + 0, lane);
    v16h b1 = load_b_frag(wlabf, nt * 2 + 1, lane);
    v8f c = {};
    c = wmma_f16(a_lbl0, b0, c);
    c = wmma_f16(a_lbl1, b1, c);
    int col = nt * 16 + nlo;
#pragma unroll
    for (int v = 0; v < 8; ++v) {
      float x = c[v];
      x = (x >= 0.0f) ? x : 0.2f * x;             // leaky_relu(0.2)
      s_lab[(v + hi * 8) * FPAD + col] = x;
    }
  }
  __syncthreads();

  v16h a_lab0 = make_a_from_lds(s_lab, nlo, hi, 0, 1.0f);
  v16h a_lab1 = make_a_from_lds(s_lab, nlo, hi, 1, 1.0f);

  // GEMM 2: thetal = lab @ Wthl + b
  {
    v16h b0 = load_b_frag(wthlf, 0, lane);
    v16h b1 = load_b_frag(wthlf, 1, lane);
    v8f c = {};
    c = wmma_f16(a_lab0, b0, c);
    c = wmma_f16(a_lab1, b1, c);
#pragma unroll
    for (int v = 0; v < 8; ++v) s_thl[(v + hi * 8) * 16 + nlo] = c[v];
  }
  __syncthreads();

  // GEMM 3 (fused): t[e][r] = (sum_f theta1[e,r,f] * x1[e,f]) * thetal[e,r]
  for (int r = 0; r < RANKN; ++r) {
    float tp[8];
#pragma unroll
    for (int v = 0; v < 8; ++v) tp[v] = 0.0f;
#pragma unroll
    for (int nt = 0; nt < 3; ++nt) {
      int slot = (r * 3 + nt) * 2;
      v16h b0 = load_b_frag(w1, slot + 0, lane);    // LDS-resident
      v16h b1 = load_b_frag(w1, slot + 1, lane);
      v8f c = {};
      c = wmma_f16(a_lab0, b0, c);
      c = wmma_f16(a_lab1, b1, c);
      int f = nt * 16 + nlo;                        // padded cols give x1==0
#pragma unroll
      for (int v = 0; v < 8; ++v) tp[v] += c[v] * s_x1[(v + hi * 8) * FPAD + f];
    }
#pragma unroll
    for (int v = 0; v < 8; ++v) {
#pragma unroll
      for (int m = 1; m <= 8; m <<= 1) tp[v] += __shfl_xor(tp[v], m, 32);
    }
    if (nlo == 0) {
#pragma unroll
      for (int v = 0; v < 8; ++v) {
        int e = v + hi * 8;
        s_t[e * RANKN + r] = tp[v] * s_thl[e * 16 + r];
      }
    }
    __syncthreads();
  }

  // GEMM 4 (fused): x[e,f] = sum_r theta2[e,r,f] * t[e,r];
  // out[n,f] = (1/K) sum_e x[e,f] * exp(-d_label[e]/10)
#pragma unroll
  for (int nt = 0; nt < 3; ++nt) {
    float acc[8];
#pragma unroll
    for (int v = 0; v < 8; ++v) acc[v] = 0.0f;
    for (int r = 0; r < RANKN; ++r) {
      int slot = (r * 3 + nt) * 2;
      v16h b0 = load_b_frag(w2, slot + 0, lane);    // LDS-resident
      v16h b1 = load_b_frag(w2, slot + 1, lane);
      v8f c = {};
      c = wmma_f16(a_lab0, b0, c);
      c = wmma_f16(a_lab1, b1, c);
#pragma unroll
      for (int v = 0; v < 8; ++v) acc[v] += c[v] * s_t[(v + hi * 8) * RANKN + r];
    }
    float s = 0.0f;
#pragma unroll
    for (int v = 0; v < 8; ++v) {
      int e = v + hi * 8;
      float w = __expf(-s_dl[e] * 0.1f) * (1.0f / (float)KNN);
      s += acc[v] * w;
    }
    s += __shfl_xor(s, 16, 32);                    // combine e halves
    int f = nt * 16 + nlo;
    if (lane < 16 && f < F_OUTN) out[(size_t)node * F_OUTN + f] = s;
  }
}

// ---------------------------------------------------------------------------
extern "C" void kernel_launch(void* const* d_in, const int* in_sizes, int n_in,
                              void* d_out, int out_size, void* d_ws, size_t ws_size,
                              hipStream_t stream) {
  const float* h     = (const float*)d_in[0];
  const float* mask  = (const float*)d_in[1];
  const float* W_lab = (const float*)d_in[2];
  const float* b_lab = (const float*)d_in[3];
  const float* W_th1 = (const float*)d_in[4];
  const float* b_th1 = (const float*)d_in[5];
  const float* W_th2 = (const float*)d_in[6];
  const float* b_th2 = (const float*)d_in[7];
  const float* W_thl = (const float*)d_in[8];
  const float* b_thl = (const float*)d_in[9];
  float* out = (float*)d_out;

  char* ws = (char*)d_ws;
  _Float16* h16   = (_Float16*)(ws + WS_H16);
  float*    sq    = (float*)   (ws + WS_SQ);
  int*      nbr   = (int*)     (ws + WS_NBR);
  int*      ctr   = (int*)     (ws + WS_CTR);
  _Float16* wlabf = (_Float16*)(ws + WS_WLABF);
  _Float16* wthlf = (_Float16*)(ws + WS_WTHLF);
  _Float16* wth1f = (_Float16*)(ws + WS_WTH1F);   // th1 || th2 contiguous
  _Float16* wth2f = (_Float16*)(ws + WS_WTH2F);

  prep_h_kernel<<<(N_NODES + 255) / 256, 256, 0, stream>>>(h, h16, sq);

  {
    int tot = 1 * 3 * 2 * 256;   // W_lab: rcount=1, ntiles=3, colsPerR=44
    pack_w_kernel<<<(tot + 255) / 256, 256, 0, stream>>>(W_lab, b_lab, 44, 1, 3, 44, wlabf);
  }
  {
    int tot = 1 * 1 * 2 * 256;   // W_thl
    pack_w_kernel<<<(tot + 255) / 256, 256, 0, stream>>>(W_thl, b_thl, 11, 1, 1, 11, wthlf);
  }
  {
    int tot = RANKN * 3 * 2 * 256;  // W_th1: [44, 484], col = r*44+f
    pack_w_kernel<<<(tot + 255) / 256, 256, 0, stream>>>(W_th1, b_th1, 484, RANKN, 3, 44, wth1f);
  }
  {
    int tot = RANKN * 3 * 2 * 256;  // W_th2
    pack_w_kernel<<<(tot + 255) / 256, 256, 0, stream>>>(W_th2, b_th2, 484, RANKN, 3, 44, wth2f);
  }

  knn_kernel<<<N_NODES / 16, 32, 0, stream>>>(h16, sq, mask, nbr, ctr);

  edge_kernel<<<N_NODES / 8, 256, SMEM_EDGE, stream>>>(h, nbr, ctr, wlabf, wthlf,
                                                       wth1f, out);
}
